// Attention_Weighted_Context_Generation_64707977282011
// MI455X (gfx1250) — compile-verified, matrix operand order checked by reading, not run on hardware
//
#include <hip/hip_runtime.h>

// ---------------------------------------------------------------------------
// ctx[m,n] = sum_k attn[m,k] * feat[n,k]   (A: [M,K] row-major, B: [N,K])
// M = K = 9216, N = 256, fp32 in/out.
// f16x3 split-precision GEMM on v_wmma_f32_16x16x32_f16 (f32 accumulate).
// ---------------------------------------------------------------------------

typedef __attribute__((ext_vector_type(16))) _Float16 v16h;
typedef __attribute__((ext_vector_type(8)))  _Float16 v8h;
typedef __attribute__((ext_vector_type(8)))  float    v8f;

constexpr int M  = 9216;   // hw
constexpr int N  = 256;    // c
constexpr int K  = 9216;   // hw
constexpr int KB = 32;     // K depth of one f16 WMMA
constexpr int ASTR = 40;   // LDS row stride (halves) for 64x32 A tiles (padded)
constexpr int BSTR = 40;   // LDS row stride (halves) for 128x32 B tiles (padded)

// Split 8 consecutive f32 into hi/lo f16 pairs (x ~= hi + lo, ~22 mantissa bits).
__device__ __forceinline__ void split8(const float* __restrict__ s, v8h& hi, v8h& lo) {
  float4 f0 = ((const float4*)s)[0];
  float4 f1 = ((const float4*)s)[1];
  float x[8] = {f0.x, f0.y, f0.z, f0.w, f1.x, f1.y, f1.z, f1.w};
#pragma unroll
  for (int i = 0; i < 8; ++i) {
    _Float16 h = (_Float16)x[i];
    hi[i] = h;
    lo[i] = (_Float16)(x[i] - (float)h);
  }
}

__device__ __forceinline__ v16h join16(v8h a, v8h b) {
  return __builtin_shufflevector(a, b, 0, 1, 2, 3, 4, 5, 6, 7,
                                       8, 9, 10, 11, 12, 13, 14, 15);
}

__global__ __launch_bounds__(256)
void awcg_gemm_f16x3(const float* __restrict__ attn,  // [M,K]
                     const float* __restrict__ feat,  // [N,K]
                     float* __restrict__ out)         // [M,N]
{
  __shared__ __align__(16) _Float16 Ah[64 * ASTR];
  __shared__ __align__(16) _Float16 Al[64 * ASTR];
  __shared__ __align__(16) _Float16 Bh[128 * BSTR];
  __shared__ __align__(16) _Float16 Bl[128 * BSTR];

  const int tid    = threadIdx.x;
  const int lane   = tid & 31;
  const int w      = tid >> 5;   // wave 0..7
  const int wm     = w & 3;      // 16-row M slice within the 64-row tile
  const int wn     = w >> 2;     // 64-col N slice within the 128-col tile
  const int half16 = lane >> 4;  // 0/1 (lane group)
  const int l16    = lane & 15;

  const int m0 = blockIdx.x * 64;
  const int n0 = blockIdx.y * 128;

  // Cooperative staging assignments.
  const int ar = tid >> 2;          // A: 64 rows, 8 f32/thread
  const int ac = (tid & 3) * 8;
  const int br = tid >> 1;          // B: 128 rows, 16 f32/thread
  const int bc = (tid & 1) * 16;

  const float* aptr = attn + (long)(m0 + ar) * K + ac;
  const float* bptr = feat + (long)(n0 + br) * K + bc;

  v8f acc[4] = {};

  // A fragment addressing (ISA 16-bit A 16x32 layout):
  //   lane = M row; elements 0..7 -> K = h*8 + e, elements 8..15 -> K = 16 + h*8 + e
  const int aoff0 = (wm * 16 + l16) * ASTR + half16 * 8;
  const int aoff1 = aoff0 + 16;

  for (int kb = 0; kb < K; kb += KB) {
    // ---- stage & split one K-block into LDS ----
    {
      v8h hi, lo;
      split8(aptr + kb, hi, lo);
      *(v8h*)(Ah + ar * ASTR + ac) = hi;
      *(v8h*)(Al + ar * ASTR + ac) = lo;

      v8h h0, l0, h1, l1;
      split8(bptr + kb, h0, l0);
      split8(bptr + kb + 8, h1, l1);
      *(v8h*)(Bh + br * BSTR + bc)     = h0;
      *(v8h*)(Bl + br * BSTR + bc)     = l0;
      *(v8h*)(Bh + br * BSTR + bc + 8) = h1;
      *(v8h*)(Bl + br * BSTR + bc + 8) = l1;
    }
    if (kb + KB < K) {
      __builtin_prefetch(aptr + kb + KB, 0, 0);  // global_prefetch_b8
      __builtin_prefetch(bptr + kb + KB, 0, 0);
    }
    __syncthreads();

    // ---- WMMA compute: D = Ahi*Bhi + Ahi*Blo + Alo*Bhi (+C) ----
    v16h Ahi = join16(*(const v8h*)(Ah + aoff0), *(const v8h*)(Ah + aoff1));
    v16h Alo = join16(*(const v8h*)(Al + aoff0), *(const v8h*)(Al + aoff1));

#pragma unroll
    for (int t = 0; t < 4; ++t) {
      const int boff0 = (wn * 64 + t * 16 + l16) * BSTR + half16 * 8;
      v16h Bhi = join16(*(const v8h*)(Bh + boff0), *(const v8h*)(Bh + boff0 + 16));
      v16h Blo = join16(*(const v8h*)(Bl + boff0), *(const v8h*)(Bl + boff0 + 16));

      acc[t] = __builtin_amdgcn_wmma_f32_16x16x32_f16(
          false, Ahi, false, Bhi, (short)0, acc[t], false, false);
      acc[t] = __builtin_amdgcn_wmma_f32_16x16x32_f16(
          false, Ahi, false, Blo, (short)0, acc[t], false, false);
      acc[t] = __builtin_amdgcn_wmma_f32_16x16x32_f16(
          false, Alo, false, Bhi, (short)0, acc[t], false, false);
    }
    __syncthreads();
  }

  // ---- epilogue: 16x16 f32 C/D layout: lane%16 = N col, VGPR r -> M = (lane/16)*8 + r
  const int gm_base = m0 + wm * 16 + half16 * 8;
  const int gn_base = n0 + wn * 64 + l16;
#pragma unroll
  for (int t = 0; t < 4; ++t) {
#pragma unroll
    for (int r = 0; r < 8; ++r) {
      out[(long)(gm_base + r) * N + (gn_base + t * 16)] = acc[t][r];
    }
  }
}

extern "C" void kernel_launch(void* const* d_in, const int* in_sizes, int n_in,
                              void* d_out, int out_size, void* d_ws, size_t ws_size,
                              hipStream_t stream) {
  (void)in_sizes; (void)n_in; (void)out_size; (void)d_ws; (void)ws_size;
  const float* attn = (const float*)d_in[0];   // weights [9216, 96, 96] -> [M,K]
  const float* feat = (const float*)d_in[1];   // cnn_feature [1,256,96,96] -> [N,K]
  float* out = (float*)d_out;                  // [M, N] (== [hw,1,1,c])

  dim3 grid(M / 64, N / 128);   // 144 x 2 workgroups
  awcg_gemm_f16x3<<<grid, 256, 0, stream>>>(attn, feat, out);
}
